// GraphSwAVModel_12489764896954
// MI455X (gfx1250) — compile-verified, matrix operand order checked by reading, not run on hardware
//
#include <hip/hip_runtime.h>
#include <hip/hip_bf16.h>

// ---------------- static config (must match reference) ----------------
#define CB   64        // graphs
#define CNPG 512       // nodes per graph
#define CN   32768     // total nodes
#define CE   524288    // total edges
#define CL   16        // tokens per node
#define CH   256       // hidden
#define CP   128       // projection
#define CK1  256
#define CK2  128

// ---------------- WMMA types ----------------
typedef __bf16        bf16_t;
typedef bf16_t        v16bf __attribute__((ext_vector_type(16)));
typedef float         v8f   __attribute__((ext_vector_type(8)));
typedef unsigned int  v4u   __attribute__((ext_vector_type(4)));

union Frag { v16bf v; v4u q[2]; };

__device__ __forceinline__ unsigned int f2bf2(float lo, float hi) {
  unsigned int ul = __float_as_uint(lo), uh = __float_as_uint(hi);
  unsigned int rl = (ul + 0x7FFFu + ((ul >> 16) & 1u)) >> 16;
  unsigned int rh = (uh + 0x7FFFu + ((uh >> 16) & 1u)) >> 16;
  return (rh << 16) | (rl & 0xFFFFu);
}

// ---------------- generic fill ----------------
__global__ void __launch_bounds__(256) fillf(float* p, float v, long n) {
  long i = (long)blockIdx.x * 256 + threadIdx.x;
  if (i < n) p[i] = v;
}

// ---------------- edge weight = mean(edge_attr, axis=1) ----------------
__global__ void __launch_bounds__(256) edge_w_k(const float* ea, float* w) {
  int e = blockIdx.x * 256 + threadIdx.x;
  w[e] = 0.5f * (ea[2 * e] + ea[2 * e + 1]);
}

// ---------------- ST encoder: masked mean of embeddings ----------------
__global__ void __launch_bounds__(256) encode_k(const int* __restrict__ tokens,
                                                const float* __restrict__ emb,
                                                float* __restrict__ x) {
  int n = blockIdx.x;            // node
  int h = threadIdx.x;           // channel
  const int* tk = tokens + n * CL;
  float s = 0.f, c = 0.f;
  for (int l = 0; l < CL; ++l) {
    int t = tk[l];               // uniform per block -> scalar load
    if (t != 0) { s += emb[(size_t)t * CH + h]; c += 1.f; }
  }
  x[(size_t)n * CH + h] = s / fmaxf(c, 1.f);
}

// ---------------- degree accumulation ----------------
__global__ void __launch_bounds__(256) deg_accum_k(const int* dst, const float* w,
                                                   float* deg) {
  int e = blockIdx.x * 256 + threadIdx.x;
  atomicAdd(&deg[dst[e]], w[e]);
}

// ---------------- WMMA GEMM: C[M,N] = A[M,K] @ W[K,N] (+bias)(+relu) ----------
// Whole bf16 W tile resident in LDS (Ncols x K, +8 k-pad -> 528B stride, bank
// conflict-free b128 fragment reads). Block = 256 threads (8 waves); grid-
// strides over 16-row M-strips; no barriers inside the K accumulation loop.
__global__ void __launch_bounds__(256)
wmma_gemm(const float* __restrict__ A, int lda,
          const float* __restrict__ W, int ldw,
          const float* __restrict__ bias,
          float* __restrict__ C, int Mrows, int Ncols, int K, int relu) {
  extern __shared__ __align__(16) unsigned short smem[];
  const int bk = K + 8;                     // padded k-stride (ushorts)
  unsigned short* lW = smem;                // [Ncols][bk]
  unsigned short* lA = smem + (size_t)Ncols * bk;   // [16][bk]

  const int tid  = threadIdx.x;
  const int lane = tid & 31;
  const int wave = tid >> 5;
  const int nTiles = Ncols >> 4;
  const bool has1 = (nTiles > 8);
  const int m16 = lane & 15;
  const int kb  = (lane < 16) ? 0 : 8;      // per-lane K sub-offset (ISA layout)
  const int Kh  = K >> 1;                   // bf16 pairs per row

  // ---- stage full W tile (fp32 -> bf16), transposed to [n][k] ----
  for (int i = tid; i < Kh * Ncols; i += 256) {
    int kk = i / Ncols;                     // pair index 0..Kh-1
    int n  = i - kk * Ncols;
    const float* wp = W + (size_t)(2 * kk) * ldw + n;
    *(unsigned int*)&lW[(size_t)n * bk + 2 * kk] = f2bf2(wp[0], wp[ldw]);
  }
  __syncthreads();

  const int Mstrips = Mrows >> 4;
  for (int s = blockIdx.x; s < Mstrips; s += gridDim.x) {
    const int row0 = s << 4;
    // ---- stage A strip 16 x K (fp32 -> bf16) ----
    for (int i = tid; i < 16 * Kh; i += 256) {
      int r  = i / Kh;
      int kk = i - r * Kh;
      const float* ap = A + (size_t)(row0 + r) * lda + 2 * kk;
      *(unsigned int*)&lA[(size_t)r * bk + 2 * kk] = f2bf2(ap[0], ap[1]);
    }
    __syncthreads();

    v8f acc0 = {0.f,0.f,0.f,0.f,0.f,0.f,0.f,0.f};
    v8f acc1 = {0.f,0.f,0.f,0.f,0.f,0.f,0.f,0.f};
    const int n0a = wave << 4;
    const int n0b = (wave + 8) << 4;
    for (int kc = 0; kc < K; kc += 32) {
      Frag a;
      a.q[0] = *(const v4u*)&lA[(size_t)m16 * bk + kc + kb];
      a.q[1] = *(const v4u*)&lA[(size_t)m16 * bk + kc + 16 + kb];
      {
        Frag b;
        b.q[0] = *(const v4u*)&lW[(size_t)(n0a + m16) * bk + kc + kb];
        b.q[1] = *(const v4u*)&lW[(size_t)(n0a + m16) * bk + kc + 16 + kb];
        acc0 = __builtin_amdgcn_wmma_f32_16x16x32_bf16(false, a.v, false, b.v,
                                                       (short)0, acc0, false, false);
      }
      if (has1) {
        Frag b;
        b.q[0] = *(const v4u*)&lW[(size_t)(n0b + m16) * bk + kc + kb];
        b.q[1] = *(const v4u*)&lW[(size_t)(n0b + m16) * bk + kc + 16 + kb];
        acc1 = __builtin_amdgcn_wmma_f32_16x16x32_bf16(false, a.v, false, b.v,
                                                       (short)0, acc1, false, false);
      }
    }

    const int mOfs = (lane < 16) ? 0 : 8;   // C/D layout: vgpr i -> M=i (+8 hi half)
    {
      int n = n0a + m16;
      float bv = bias ? bias[n] : 0.f;
      for (int i2 = 0; i2 < 8; ++i2) {
        float v = acc0[i2] + bv;
        if (relu) v = fmaxf(v, 0.f);
        C[(size_t)(row0 + i2 + mOfs) * Ncols + n] = v;
      }
    }
    if (has1) {
      int n = n0b + m16;
      float bv = bias ? bias[n] : 0.f;
      for (int i2 = 0; i2 < 8; ++i2) {
        float v = acc1[i2] + bv;
        if (relu) v = fmaxf(v, 0.f);
        C[(size_t)(row0 + i2 + mOfs) * Ncols + n] = v;
      }
    }
    __syncthreads();   // protect lA before next strip restage
  }
}

// ---------------- edge aggregation: agg[dst] += h[src]*norm ----------------
__global__ void __launch_bounds__(256) edge_agg_k(const int* __restrict__ src,
                                                  const int* __restrict__ dst,
                                                  const float* __restrict__ w,
                                                  const float* __restrict__ deg,
                                                  const float* __restrict__ hm,
                                                  float* __restrict__ agg) {
  int e = blockIdx.x;
  float wv = w[e];
  if (wv == 0.f) return;                 // uniform per block
  int s = src[e], d = dst[e];
  float norm = rsqrtf(deg[s]) * wv * rsqrtf(deg[d]);
  int t = threadIdx.x;
  atomicAdd(&agg[(size_t)d * CH + t], hm[(size_t)s * CH + t] * norm);
}

// ---------------- GCN finalize: relu(agg + h/deg + b) ----------------
__global__ void __launch_bounds__(256) gcn_fin_k(const float* __restrict__ agg,
                                                 const float* __restrict__ hm,
                                                 const float* __restrict__ deg,
                                                 const float* __restrict__ b,
                                                 float* __restrict__ xo) {
  size_t i = (size_t)blockIdx.x * 256 + threadIdx.x;
  int row = (int)(i >> 8), col = (int)(i & 255);
  float v = agg[i] + hm[i] * (1.f / deg[row]) + b[col];
  xo[i] = fmaxf(v, 0.f);
}

// ---------------- TopK pooling (per-graph block, bitonic sort in LDS) --------
__global__ void __launch_bounds__(256) topk_pool_k(const float* __restrict__ x,
                                                   const float* __restrict__ p,
                                                   float* __restrict__ xnew,
                                                   int* __restrict__ nmap,
                                                   int nper, int kkeep) {
  int g = blockIdx.x;
  int tid = threadIdx.x, lane = tid & 31, wave = tid >> 5;
  __shared__ float sc[512];
  __shared__ int   idx[512];
  __shared__ int   map[512];
  // ||p|| (computed redundantly per wave)
  float pp = 0.f;
  for (int j = 0; j < 8; ++j) { float pv = p[lane + 32 * j]; pp += pv * pv; }
  for (int off = 16; off; off >>= 1) pp += __shfl_xor(pp, off, 32);
  float invn = rsqrtf(fmaxf(pp, 1e-30f));
  const float* xg = x + (size_t)g * nper * CH;
  for (int r = wave; r < nper; r += 8) {
    const float* row = xg + (size_t)r * CH;
    float s = 0.f;
    for (int j = 0; j < 8; ++j) s += row[lane + 32 * j] * p[lane + 32 * j];
    for (int off = 16; off; off >>= 1) s += __shfl_xor(s, off, 32);
    if (lane == 0) { sc[r] = s * invn; idx[r] = r; }
  }
  __syncthreads();
  // bitonic sort descending on sc/idx
  for (int kk = 2; kk <= nper; kk <<= 1) {
    for (int j = kk >> 1; j > 0; j >>= 1) {
      for (int i = tid; i < nper; i += 256) {
        int ixj = i ^ j;
        if (ixj > i) {
          bool desc = ((i & kk) == 0);
          bool sw = desc ? (sc[i] < sc[ixj]) : (sc[i] > sc[ixj]);
          if (sw) {
            float ts = sc[i]; sc[i] = sc[ixj]; sc[ixj] = ts;
            int   ti = idx[i]; idx[i] = idx[ixj]; idx[ixj] = ti;
          }
        }
      }
      __syncthreads();
    }
  }
  // old->new map
  for (int i = tid; i < nper; i += 256) map[i] = -1;
  __syncthreads();
  for (int j = tid; j < kkeep; j += 256) map[idx[j]] = g * kkeep + j;
  __syncthreads();
  for (int i = tid; i < nper; i += 256) nmap[g * nper + i] = map[i];
  // gated copy of kept rows
  int total = kkeep * CH;
  for (int e = tid; e < total; e += 256) {
    int j = e >> 8, h = e & 255;
    float gt = tanhf(sc[j]);
    xnew[((size_t)g * kkeep + j) * CH + h] = xg[(size_t)idx[j] * CH + h] * gt;
  }
}

// ---------------- edge remap after pooling ----------------
__global__ void __launch_bounds__(256) edge_remap_k(const int* src, const int* dst,
                                                    const float* w, const int* nmap,
                                                    int* src1, int* dst1, float* w1) {
  int e = blockIdx.x * 256 + threadIdx.x;
  int ns = nmap[src[e]], nd = nmap[dst[e]];
  bool keep = (ns >= 0) && (nd >= 0);
  src1[e] = keep ? ns : 0;
  dst1[e] = keep ? nd : 0;
  w1[e]   = keep ? w[e] : 0.f;
}

// ---------------- global attention pool (accumulating) ----------------
__global__ void __launch_bounds__(256) attpool_k(const float* __restrict__ x,
                                                 const float* __restrict__ wg,
                                                 const float* __restrict__ bg,
                                                 float* __restrict__ out, int k) {
  int g = blockIdx.x, tid = threadIdx.x, lane = tid & 31, wave = tid >> 5;
  __shared__ float gt[256];
  __shared__ float red[256];
  const float* xg = x + (size_t)g * k * CH;
  for (int r = wave; r < k; r += 8) {
    const float* row = xg + (size_t)r * CH;
    float s = 0.f;
    for (int j = 0; j < 8; ++j) s += row[lane + 32 * j] * wg[lane + 32 * j];
    for (int off = 16; off; off >>= 1) s += __shfl_xor(s, off, 32);
    if (lane == 0) gt[r] = s + bg[0];
  }
  __syncthreads();
  red[tid] = (tid < k) ? gt[tid] : -3.4e38f;
  __syncthreads();
  for (int s = 128; s > 0; s >>= 1) { if (tid < s) red[tid] = fmaxf(red[tid], red[tid + s]); __syncthreads(); }
  float mx = red[0];
  __syncthreads();
  float ev = (tid < k) ? expf(gt[tid] - mx) : 0.f;
  if (tid < k) gt[tid] = ev;
  red[tid] = ev;
  __syncthreads();
  for (int s = 128; s > 0; s >>= 1) { if (tid < s) red[tid] += red[tid + s]; __syncthreads(); }
  float inv = 1.f / red[0];
  __syncthreads();
  float acc = 0.f;
  for (int i = 0; i < k; ++i) acc += gt[i] * xg[(size_t)i * CH + tid];
  out[(size_t)g * CH + tid] += acc * inv;
}

// ---------------- final: L2-normalized logits + copy 'out' ----------------
__global__ void __launch_bounds__(256) finalize_k(const float* __restrict__ lg,
                                                  const float* __restrict__ outb,
                                                  float* __restrict__ dout) {
  int r = blockIdx.x, t = threadIdx.x;
  __shared__ float red[256];
  float v = (t < CP) ? lg[r * CP + t] : 0.f;
  red[t] = v * v;
  __syncthreads();
  for (int s = 128; s > 0; s >>= 1) { if (t < s) red[t] += red[t + s]; __syncthreads(); }
  float inv = 1.f / fmaxf(sqrtf(red[0]), 1e-12f);
  if (t < CP) dout[r * CP + t] = v * inv;
  dout[CB * CP + r * CH + t] = outb[r * CH + t];
}

// ---------------- host-side orchestration ----------------
static inline void launch_gemm(const float* A, int lda, const float* W, int ldw,
                               const float* bias, float* C, int M, int Ncols,
                               int K, int relu, hipStream_t stream) {
  int strips = M >> 4;
  int grid = strips < 256 ? strips : 256;
  size_t shbytes = (size_t)(Ncols + 16) * (K + 8) * sizeof(unsigned short);
  wmma_gemm<<<grid, 256, shbytes, stream>>>(A, lda, W, ldw, bias, C, M, Ncols, K, relu);
}

extern "C" void kernel_launch(void* const* d_in, const int* in_sizes, int n_in,
                              void* d_out, int out_size, void* d_ws, size_t ws_size,
                              hipStream_t stream) {
  (void)in_sizes; (void)n_in; (void)out_size; (void)ws_size;
  const int*   tokens = (const int*)d_in[0];
  const int*   src    = (const int*)d_in[1];
  const int*   dst    = (const int*)d_in[2];
  const float* eattr  = (const float*)d_in[3];
  const float* emb    = (const float*)d_in[4];
  const float* W0 = (const float*)d_in[5];  const float* b0  = (const float*)d_in[6];
  const float* W1 = (const float*)d_in[7];  const float* b1  = (const float*)d_in[8];
  const float* p0 = (const float*)d_in[9];  const float* p1  = (const float*)d_in[10];
  const float* wg = (const float*)d_in[11]; const float* bg  = (const float*)d_in[12];
  const float* Wp1 = (const float*)d_in[13]; const float* bp1 = (const float*)d_in[14];
  const float* Wp2 = (const float*)d_in[15]; const float* bp2 = (const float*)d_in[16];
  float* dout = (float*)d_out;

  // workspace layout
  char* wsb = (char*)d_ws;
  float* w    = (float*)wsb;  wsb += (size_t)CE * 4;
  float* x    = (float*)wsb;  wsb += (size_t)CN * CH * 4;   // encoder out / GCN1 out
  float* hbuf = (float*)wsb;  wsb += (size_t)CN * CH * 4;   // x@W (both layers)
  float* agg  = (float*)wsb;  wsb += (size_t)CN * CH * 4;   // scatter target
  float* deg  = (float*)wsb;  wsb += (size_t)CN * 4;
  float* x1   = (float*)wsb;  wsb += (size_t)CB * CK1 * CH * 4; // pool1 out / GCN2 out
  int*   nmap = (int*)wsb;    wsb += (size_t)CN * 4;
  int*   src1 = (int*)wsb;    wsb += (size_t)CE * 4;
  int*   dst1 = (int*)wsb;    wsb += (size_t)CE * 4;
  float* w1   = (float*)wsb;  wsb += (size_t)CE * 4;
  float* x2   = (float*)wsb;  wsb += (size_t)CB * CK2 * CH * 4;
  float* outb = (float*)wsb;  wsb += (size_t)CB * CH * 4;
  float* p1b  = (float*)wsb;  wsb += (size_t)CB * CP * 4;
  float* lgb  = (float*)wsb;  wsb += (size_t)CB * CP * 4;

  // edge weights + encoder
  edge_w_k<<<CE / 256, 256, 0, stream>>>(eattr, w);
  encode_k<<<CN, 256, 0, stream>>>(tokens, emb, x);

  // ---- GCN layer 1 ----
  fillf<<<(CN + 255) / 256, 256, 0, stream>>>(deg, 1.0f, CN);
  deg_accum_k<<<CE / 256, 256, 0, stream>>>(dst, w, deg);
  launch_gemm(x, CH, W0, CH, nullptr, hbuf, CN, CH, CH, 0, stream);
  fillf<<<(int)(((size_t)CN * CH + 255) / 256), 256, 0, stream>>>(agg, 0.f, (long)CN * CH);
  edge_agg_k<<<CE, 256, 0, stream>>>(src, dst, w, deg, hbuf, agg);
  gcn_fin_k<<<(int)(((size_t)CN * CH) / 256), 256, 0, stream>>>(agg, hbuf, deg, b0, x);

  // ---- pool 1 + attpool 1 ----
  topk_pool_k<<<CB, 256, 0, stream>>>(x, p0, x1, nmap, CNPG, CK1);
  edge_remap_k<<<CE / 256, 256, 0, stream>>>(src, dst, w, nmap, src1, dst1, w1);
  fillf<<<(CB * CH + 255) / 256, 256, 0, stream>>>(outb, 0.f, CB * CH);
  attpool_k<<<CB, 256, 0, stream>>>(x1, wg, bg, outb, CK1);

  // ---- GCN layer 2 (on 16384 pooled nodes) ----
  const int N2 = CB * CK1;
  fillf<<<(N2 + 255) / 256, 256, 0, stream>>>(deg, 1.0f, N2);
  deg_accum_k<<<CE / 256, 256, 0, stream>>>(dst1, w1, deg);
  launch_gemm(x1, CH, W1, CH, nullptr, hbuf, N2, CH, CH, 0, stream);
  fillf<<<(int)(((size_t)N2 * CH + 255) / 256), 256, 0, stream>>>(agg, 0.f, (long)N2 * CH);
  edge_agg_k<<<CE, 256, 0, stream>>>(src1, dst1, w1, deg, hbuf, agg);
  gcn_fin_k<<<(int)(((size_t)N2 * CH) / 256), 256, 0, stream>>>(agg, hbuf, deg, b1, x1);

  // ---- pool 2 + attpool 2 (edges not needed after this) ----
  topk_pool_k<<<CB, 256, 0, stream>>>(x1, p1, x2, nmap, CK1, CK2);
  attpool_k<<<CB, 256, 0, stream>>>(x2, wg, bg, outb, CK2);

  // ---- projection head ----
  launch_gemm(outb, CH, Wp1, CP, bp1, p1b, CB, CP, CH, 1, stream);
  launch_gemm(p1b, CP, Wp2, CP, bp2, lgb, CB, CP, CP, 0, stream);
  finalize_k<<<CB, 256, 0, stream>>>(lgb, outb, dout);
}
